// MultiHeadSelfAttention_47132971106473
// MI455X (gfx1250) — compile-verified, matrix operand order checked by reading, not run on hardware
//
#include <hip/hip_runtime.h>

// ---------------------------------------------------------------------------
// MHA forward for MI455X (gfx1250): bf16 WMMA everywhere, fp32 accumulate.
// L=2048, N=2, E=2048, H=16, DH=128.
// GEMMs use async global->LDS staging (ASYNCcnt) with double buffering.
// ---------------------------------------------------------------------------

#define LSEQ  2048
#define NB    2
#define EMB   2048
#define NH    16
#define DH    128
#define M_ROWS (LSEQ * NB)          // 4096
#define QKV_COLS (3 * EMB)          // 6144
#define SCALE_F 0.08838834764831845f // 1/sqrt(128)

typedef __attribute__((ext_vector_type(16))) __bf16 bf16x16;
typedef __attribute__((ext_vector_type(8)))  float  f32x8;
typedef unsigned short u16;

__device__ __forceinline__ u16 f2bf(float f) {
    union { float f; unsigned u; } v; v.f = f;
    unsigned r = v.u + 0x7FFFu + ((v.u >> 16) & 1u);   // round-to-nearest-even
    return (u16)(r >> 16);
}

// Async copy of 16 bytes global -> LDS (CDNA5, tracked by ASYNCcnt).
// lds_addr: per-thread LDS byte address (low 32 bits of generic pointer).
__device__ __forceinline__ void async_b128(unsigned lds_addr, const u16* gaddr) {
    asm volatile("global_load_async_to_lds_b128 %0, %1, off"
                 :: "v"(lds_addr), "v"(gaddr) : "memory");
}

// ---------------------------------------------------------------------------
// fp32 -> bf16 bulk convert
// ---------------------------------------------------------------------------
__global__ void cvt_f32_bf16_kernel(const float* __restrict__ in,
                                    u16* __restrict__ out, long n) {
    long i = (long)blockIdx.x * blockDim.x + threadIdx.x;
    if (i < n) out[i] = f2bf(in[i]);
}

// ---------------------------------------------------------------------------
// Split fp32 QKV [4096][6144] into bf16 Q/K ([nh][L][DH], Q pre-scaled) and
// V transposed ([nh][DH][L]) so attention fragment loads are contiguous.
// ---------------------------------------------------------------------------
__global__ void qkv_split_kernel(const float* __restrict__ qkv,
                                 u16* __restrict__ Qb,
                                 u16* __restrict__ Kb,
                                 u16* __restrict__ Vt) {
    long i = (long)blockIdx.x * blockDim.x + threadIdx.x;
    const long total = (long)M_ROWS * QKV_COLS;
    if (i >= total) return;
    int col  = (int)(i % QKV_COLS);
    int row  = (int)(i / QKV_COLS);   // l*NB + n
    int l    = row >> 1;
    int n    = row & 1;
    int part = col >> 11;             // 0=Q 1=K 2=V
    int e    = col & (EMB - 1);
    int h    = e >> 7;
    int dh   = e & (DH - 1);
    int nh   = n * NH + h;
    float v  = qkv[i];
    if (part == 0)
        Qb[((long)nh * LSEQ + l) * DH + dh] = f2bf(v * SCALE_F);
    else if (part == 1)
        Kb[((long)nh * LSEQ + l) * DH + dh] = f2bf(v);
    else
        Vt[((long)nh * DH + dh) * LSEQ + l] = f2bf(v);
}

// ---------------------------------------------------------------------------
// C[M][Nd] = A[M][K] * B[Nd][K]^T   (bf16 in, fp32 out), WMMA 16x16x32 bf16.
// Workgroup tile 128x128, 8 waves, each wave 32x64 (2x4 WMMA accumulators).
// Double-buffered LDS (32 KB) fed by global_load_async_to_lds_b128; the async
// engine streams tile k+1 while WMMAs consume tile k (s_wait_asynccnt).
// ---------------------------------------------------------------------------
#define TM 128
#define TN 128
#define TK 32

__global__ __launch_bounds__(256) void gemm_bf16_wmma_kernel(
        const u16* __restrict__ A, const u16* __restrict__ B,
        float* __restrict__ C, int M, int Nd, int K) {
    __shared__ __align__(64) u16 lA[2][TM * TK];   // 2 x 8 KB
    __shared__ __align__(64) u16 lB[2][TN * TK];   // 2 x 8 KB

    const int tile_m = blockIdx.y * TM;
    const int tile_n = blockIdx.x * TN;
    const int tid  = threadIdx.x;
    const int wave = tid >> 5;
    const int lane = tid & 31;
    const int wm = (wave & 3) * 32;     // 4 waves down
    const int wn = (wave >> 2) * 64;    // 2 waves across
    const int lr = lane & 15;
    const int kh = (lane >> 4) * 16;    // K-half select for fragments

    // per-thread staging chunks: tile is 128x32 bf16 = 512 uint4; 2 per thread
    const int cr0 = tid >> 2, cc0 = (tid & 3) * 8;          // chunk rows 0..63
    const int cr1 = (tid + 256) >> 2, cc1 = cc0;            // chunk rows 64..127

    f32x8 acc[2][4] = {};

    auto issue_tile = [&](int k0, int buf) {
        async_b128((unsigned)(uintptr_t)&lA[buf][cr0 * TK + cc0],
                   &A[(long)(tile_m + cr0) * K + k0 + cc0]);
        async_b128((unsigned)(uintptr_t)&lA[buf][cr1 * TK + cc1],
                   &A[(long)(tile_m + cr1) * K + k0 + cc1]);
        async_b128((unsigned)(uintptr_t)&lB[buf][cr0 * TK + cc0],
                   &B[(long)(tile_n + cr0) * K + k0 + cc0]);
        async_b128((unsigned)(uintptr_t)&lB[buf][cr1 * TK + cc1],
                   &B[(long)(tile_n + cr1) * K + k0 + cc1]);
    };

    issue_tile(0, 0);
    const int nk = K / TK;
    for (int kt = 0; kt < nk; ++kt) {
        const int buf = kt & 1;
        if (kt + 1 < nk) {
            issue_tile((kt + 1) * TK, buf ^ 1);
            // 8 outstanding per wave; <=4 means this buffer's 4 have landed
            asm volatile("s_wait_asynccnt 0x4" ::: "memory");
        } else {
            asm volatile("s_wait_asynccnt 0x0" ::: "memory");
        }
        __syncthreads();   // all waves' portions of buf are visible

        bf16x16 af[2], bfr[4];
        #pragma unroll
        for (int i = 0; i < 2; ++i)
            af[i] = *(const bf16x16*)&lA[buf][(wm + i * 16 + lr) * TK + kh];
        #pragma unroll
        for (int i = 0; i < 4; ++i)
            bfr[i] = *(const bf16x16*)&lB[buf][(wn + i * 16 + lr) * TK + kh];

        #pragma unroll
        for (int mi = 0; mi < 2; ++mi)
            #pragma unroll
            for (int ni = 0; ni < 4; ++ni)
                acc[mi][ni] = __builtin_amdgcn_wmma_f32_16x16x32_bf16(
                    false, af[mi], false, bfr[ni], (short)0, acc[mi][ni],
                    false, false);

        __syncthreads();   // protect buf from being overwritten next round
    }

    // C/D layout: n = lane%16 ; m = r + 8*(lane>>4)
    const int mh = (lane >> 4) * 8;
    #pragma unroll
    for (int mi = 0; mi < 2; ++mi)
        #pragma unroll
        for (int ni = 0; ni < 4; ++ni)
            #pragma unroll
            for (int r = 0; r < 8; ++r) {
                int row = tile_m + wm + mi * 16 + r + mh;
                int col = tile_n + wn + ni * 16 + lr;
                C[(long)row * Nd + col] = acc[mi][ni][r];
            }
}

// ---------------------------------------------------------------------------
// Causal flash attention. Grid: (L/128, NB*NH). 256 threads = 8 waves.
// Each wave owns 16 query rows; streams 32-key blocks with online softmax.
// ---------------------------------------------------------------------------
__global__ __launch_bounds__(256) void flash_attn_kernel(
        const u16* __restrict__ Qb,   // [nh][L][DH]
        const u16* __restrict__ Kb,   // [nh][L][DH]
        const u16* __restrict__ Vt,   // [nh][DH][L]
        float* __restrict__ Ob) {     // [L*NB][EMB]
    const int nh   = blockIdx.y;
    const int n    = nh / NH;
    const int h    = nh % NH;
    const int wave = threadIdx.x >> 5;
    const int lane = threadIdx.x & 31;
    const int lr   = lane & 15;
    const int kh   = (lane >> 4) * 16;
    const int mh   = (lane >> 4) * 8;
    const int qrow0 = blockIdx.x * 128 + wave * 16;

    const u16* Q  = Qb + (long)nh * LSEQ * DH;
    const u16* Kp = Kb + (long)nh * LSEQ * DH;
    const u16* Vp = Vt + (long)nh * DH * LSEQ;

    // resident Q fragments (16 rows x 128 dh = 4 K-chunks)
    bf16x16 qf[4];
    #pragma unroll
    for (int c = 0; c < 4; ++c)
        qf[c] = *(const bf16x16*)&Q[(long)(qrow0 + lr) * DH + c * 32 + kh];

    f32x8 o_acc[8] = {};
    float mrow[8], lrow[8];
    #pragma unroll
    for (int r = 0; r < 8; ++r) { mrow[r] = -3.0e38f; lrow[r] = 0.0f; }

    __shared__ __align__(64) u16 pbuf[8][16 * 32];   // per-wave P tile (8 KB)
    u16* pb = pbuf[wave];

    const int nkb = (qrow0 >> 5) + 1;   // 32-key blocks covering keys <= qrow0+15
    for (int kb = 0; kb < nkb; ++kb) {
        const int key0 = kb * 32;

        // S = Q K^T for 16 queries x 32 keys (two 16-wide WMMA columns)
        f32x8 s0 = {}, s1 = {};
        #pragma unroll
        for (int c = 0; c < 4; ++c) {
            bf16x16 k0f = *(const bf16x16*)&Kp[(long)(key0 + lr) * DH + c * 32 + kh];
            bf16x16 k1f = *(const bf16x16*)&Kp[(long)(key0 + 16 + lr) * DH + c * 32 + kh];
            s0 = __builtin_amdgcn_wmma_f32_16x16x32_bf16(false, qf[c], false, k0f,
                                                         (short)0, s0, false, false);
            s1 = __builtin_amdgcn_wmma_f32_16x16x32_bf16(false, qf[c], false, k1f,
                                                         (short)0, s1, false, false);
        }

        // causal mask + online softmax (row reductions across 16 lanes/half)
        float p0[8], p1[8];
        #pragma unroll
        for (int r = 0; r < 8; ++r) {
            int mg = qrow0 + mh + r;
            float v0 = (key0 + lr      <= mg) ? s0[r] : -3.0e38f;
            float v1 = (key0 + 16 + lr <= mg) ? s1[r] : -3.0e38f;
            float rx = fmaxf(v0, v1);
            #pragma unroll
            for (int off = 8; off >= 1; off >>= 1)
                rx = fmaxf(rx, __shfl_xor(rx, off, 32));
            float mnew = fmaxf(mrow[r], rx);
            float e0 = __expf(v0 - mnew);
            float e1 = __expf(v1 - mnew);
            float corr = __expf(mrow[r] - mnew);
            float rs = e0 + e1;
            #pragma unroll
            for (int off = 8; off >= 1; off >>= 1)
                rs += __shfl_xor(rs, off, 32);
            lrow[r] = lrow[r] * corr + rs;
            mrow[r] = mnew;
            p0[r] = e0; p1[r] = e1;
            #pragma unroll
            for (int d = 0; d < 8; ++d)
                o_acc[d][r] *= corr;
        }

        // round-trip P through per-wave LDS to reach A-fragment layout
        #pragma unroll
        for (int r = 0; r < 8; ++r) {
            int m = r + mh;
            pb[m * 32 + lr]      = f2bf(p0[r]);
            pb[m * 32 + 16 + lr] = f2bf(p1[r]);
        }
        asm volatile("s_wait_dscnt 0x0" ::: "memory");  // DS in-order per wave
        bf16x16 pf = *(const bf16x16*)&pb[lr * 32 + kh];

        // O += P V  (8 dh-chunks of 16)
        #pragma unroll
        for (int d = 0; d < 8; ++d) {
            bf16x16 vf = *(const bf16x16*)&Vp[(long)(d * 16 + lr) * LSEQ + key0 + kh];
            o_acc[d] = __builtin_amdgcn_wmma_f32_16x16x32_bf16(
                false, pf, false, vf, (short)0, o_acc[d], false, false);
        }
    }

    // normalize and scatter to [l*NB + n][h*DH + dh]
    #pragma unroll
    for (int r = 0; r < 8; ++r) {
        float inv = 1.0f / lrow[r];
        int l = qrow0 + mh + r;
        long row = (long)(l * NB + n) * EMB + h * DH;
        #pragma unroll
        for (int d = 0; d < 8; ++d)
            Ob[row + d * 16 + lr] = o_acc[d][r] * inv;
    }
}

// ---------------------------------------------------------------------------
// Host-side orchestration
// ---------------------------------------------------------------------------
extern "C" void kernel_launch(void* const* d_in, const int* in_sizes, int n_in,
                              void* d_out, int out_size, void* d_ws, size_t ws_size,
                              hipStream_t stream) {
    const float* x    = (const float*)d_in[0];   // [L,N,E]
    const float* wqkv = (const float*)d_in[1];   // [3E,E]
    const float* wout = (const float*)d_in[2];   // [E,E]
    float* out = (float*)d_out;                  // [L,N,E]

    char* p = (char*)d_ws;
    auto take = [&](size_t bytes) {
        char* r = p;
        p += (bytes + 255) & ~(size_t)255;
        return r;
    };

    u16*   Xb    = (u16*)  take((size_t)M_ROWS * EMB * 2);
    u16*   Wqb   = (u16*)  take((size_t)QKV_COLS * EMB * 2);
    u16*   Wob   = (u16*)  take((size_t)EMB * EMB * 2);
    float* QKV   = (float*)take((size_t)M_ROWS * QKV_COLS * 4);
    u16*   Qb    = (u16*)  take((size_t)NB * NH * LSEQ * DH * 2);
    u16*   Kb    = (u16*)  take((size_t)NB * NH * LSEQ * DH * 2);
    u16*   Vt    = (u16*)  take((size_t)NB * NH * DH * LSEQ * 2);
    float* Attn  = (float*)take((size_t)M_ROWS * EMB * 4);
    u16*   Attnb = (u16*)  take((size_t)M_ROWS * EMB * 2);

    const long nx = (long)M_ROWS * EMB;
    const long nq = (long)QKV_COLS * EMB;
    const long no = (long)EMB * EMB;
    cvt_f32_bf16_kernel<<<dim3((nx + 255) / 256), dim3(256), 0, stream>>>(x, Xb, nx);
    cvt_f32_bf16_kernel<<<dim3((nq + 255) / 256), dim3(256), 0, stream>>>(wqkv, Wqb, nq);
    cvt_f32_bf16_kernel<<<dim3((no + 255) / 256), dim3(256), 0, stream>>>(wout, Wob, no);

    // QKV projection: [4096 x 2048] * [6144 x 2048]^T -> [4096 x 6144]
    gemm_bf16_wmma_kernel<<<dim3(QKV_COLS / TN, M_ROWS / TM), dim3(256), 0, stream>>>(
        Xb, Wqb, QKV, M_ROWS, QKV_COLS, EMB);

    const long nsplit = (long)M_ROWS * QKV_COLS;
    qkv_split_kernel<<<dim3((nsplit + 255) / 256), dim3(256), 0, stream>>>(
        QKV, Qb, Kb, Vt);

    flash_attn_kernel<<<dim3(LSEQ / 128, NB * NH), dim3(256), 0, stream>>>(
        Qb, Kb, Vt, Attn);

    cvt_f32_bf16_kernel<<<dim3((nx + 255) / 256), dim3(256), 0, stream>>>(Attn, Attnb, nx);

    // output projection: [4096 x 2048] * [2048 x 2048]^T -> d_out
    gemm_bf16_wmma_kernel<<<dim3(EMB / TN, M_ROWS / TM), dim3(256), 0, stream>>>(
        Attnb, Wob, out, M_ROWS, EMB, EMB);
}
